// IFHMM_86852828660373
// MI455X (gfx1250) — matrix operand order repaired and checked by reading, not compiled, not run on hardware
//
#include <hip/hip_runtime.h>
#include <hip/hip_bf16.h>
#include <math.h>

typedef __attribute__((ext_vector_type(16))) _Float16 v16h;
typedef __attribute__((ext_vector_type(8)))  _Float16 h8;
typedef __attribute__((ext_vector_type(8)))  float    v8f;
typedef __attribute__((ext_vector_type(4)))  float    f4;   // native vec for NT stores

// Problem constants (fixed by the reference harness).
#define S_   256
#define C_   4
#define N_   1000
#define T_   200
#define NP_  1008   // N padded to 63*16 for WMMA M-tiles
#define TC_  800    // T*C columns (50*16 WMMA N-tiles)

// Output float offsets (tuple concatenated flat, in return order).
#define OUT_OBS   0
#define OUT_OBSF  (OUT_OBS  + (size_t)T_*N_)          // log_obs_  [T,S,C,N]
#define OUT_HID   (OUT_OBSF + (size_t)T_*S_*C_*N_)    // log_hidden[T,S,C]
#define OUT_EMI   (OUT_HID  + (size_t)T_*S_*C_)       // log_emission[S,N]
#define OUT_W     (OUT_EMI  + (size_t)S_*N_)          // log_chain_weights[C]

// Workspace byte offsets (all 256B aligned).
// NOTE: expEt / expHt are stored S-minor (transposed) so each WMMA lane's
// K-run is contiguous -> fragments load as global_load_b128.
#define WS_EXPTTR 0                                    // f32 [256][256] exp(T)^T
#define WS_EXPET  (WS_EXPTTR + 256*256*4)              // f16 [1008][256] exp(E)^T, pad rows 0
#define WS_EXPHT  (WS_EXPET  + 1008*256*2)             // f16 [800][256]  exp(h)^T, row = t*4+c
#define WS_WEXP   (WS_EXPHT  + 800*256*2)              // f32 [4] exp(log_chain_weights)

// ---------------------------------------------------------------------------
// Small normalizations: chain weights + state-init column log-softmax.
// Seeds h_0 into log_hidden, exp(h_0) into expHt rows 0..3, zeroes expEt pad.
__global__ void ifhmm_norm_small(const float* __restrict__ usi,
                                 const float* __restrict__ ucw,
                                 float* __restrict__ out_hid,
                                 float* __restrict__ out_w,
                                 float* __restrict__ wexp,
                                 _Float16* __restrict__ expHt,
                                 _Float16* __restrict__ expEt) {
    __shared__ float lse[C_];
    int tid = threadIdx.x;
    if (tid == 0) {
        float m = ucw[0];
        for (int c = 1; c < C_; ++c) m = fmaxf(m, ucw[c]);
        float s = 0.f;
        for (int c = 0; c < C_; ++c) s += expf(ucw[c] - m);
        float l = m + logf(s);
        for (int c = 0; c < C_; ++c) {
            float v = ucw[c] - l;
            out_w[c] = v;
            wexp[c]  = expf(v);
        }
    }
    if (tid < C_) {
        int c = tid;
        float m = -__builtin_inff();
        for (int s = 0; s < S_; ++s) m = fmaxf(m, usi[s * C_ + c]);
        float sum = 0.f;
        for (int s = 0; s < S_; ++s) sum += expf(usi[s * C_ + c] - m);
        lse[c] = m + logf(sum);
    }
    __syncthreads();
    for (int i = tid; i < S_ * C_; i += blockDim.x) {
        float h0 = usi[i] - lse[i & 3];
        out_hid[i] = h0;                                  // t = 0
        expHt[(i & 3) * S_ + (i >> 2)] = (_Float16)expf(h0);
    }
    // Zero the 8 padded n-rows of expEt so WMMA never sees poison.
    for (int i = tid; i < 8 * S_; i += blockDim.x)
        expEt[N_ * S_ + i] = (_Float16)0.f;
}

// ---------------------------------------------------------------------------
// Transition row log-softmax -> exp, stored transposed: expTtr[s'][s].
__global__ void ifhmm_norm_transition(const float* __restrict__ ut,
                                      float* __restrict__ expTtr) {
    __shared__ float red[256];
    int r = blockIdx.x, tid = threadIdx.x;
    float x = ut[r * S_ + tid];
    red[tid] = x;
    __syncthreads();
    for (int off = 128; off > 0; off >>= 1) {
        if (tid < off) red[tid] = fmaxf(red[tid], red[tid + off]);
        __syncthreads();
    }
    float m = red[0];
    __syncthreads();
    red[tid] = expf(x - m);
    __syncthreads();
    for (int off = 128; off > 0; off >>= 1) {
        if (tid < off) red[tid] += red[tid + off];
        __syncthreads();
    }
    float lse = m + logf(red[0]);
    expTtr[tid * S_ + r] = expf(x - lse);
}

// ---------------------------------------------------------------------------
// Emission row log-softmax over N=1000: normalized rows to d_out, exp() as
// f16 into expEt TRANSPOSED ([n][s]) for contiguous WMMA K-runs.
__global__ void ifhmm_norm_emission(const float* __restrict__ ue,
                                    float* __restrict__ out_emi,
                                    _Float16* __restrict__ expEt) {
    __shared__ float red[256];
    int s = blockIdx.x, tid = threadIdx.x;
    const float* row = ue + (size_t)s * N_;
    float m = -__builtin_inff();
    for (int n = tid; n < N_; n += 256) m = fmaxf(m, row[n]);
    red[tid] = m;
    __syncthreads();
    for (int off = 128; off > 0; off >>= 1) {
        if (tid < off) red[tid] = fmaxf(red[tid], red[tid + off]);
        __syncthreads();
    }
    m = red[0];
    __syncthreads();
    float sum = 0.f;
    for (int n = tid; n < N_; n += 256) sum += expf(row[n] - m);
    red[tid] = sum;
    __syncthreads();
    for (int off = 128; off > 0; off >>= 1) {
        if (tid < off) red[tid] += red[tid + off];
        __syncthreads();
    }
    float lse = m + logf(red[0]);
    for (int n = tid; n < N_; n += 256) {
        float v = row[n] - lse;
        out_emi[(size_t)s * N_ + n] = v;
        expEt[n * S_ + s] = (_Float16)expf(v);
    }
}

// ---------------------------------------------------------------------------
// Sequential forward recursion, one WGP (1024 threads = 32 waves).
// exp-space: p[s',c] = sum_s expT'[s'][s] * eh[s][c];  h' = log(p); eh' = p.
// eh kept in LDS, c-major with stride 260 (bank-conflict-free), float2 MACs.
__global__ void ifhmm_forward_recursion(const float* __restrict__ expTtr,
                                        float* __restrict__ out_hid,
                                        _Float16* __restrict__ expHt) {
    __shared__ float ehc[C_ * 260];
    int tid = threadIdx.x;          // 1024
    int sp = tid >> 2, c = tid & 3;
    ehc[c * 260 + sp] = expf(out_hid[tid]);
    __syncthreads();
    const float* trow = expTtr + sp * S_;
    const float* eh   = &ehc[c * 260];
    for (int t = 1; t < T_; ++t) {
        float2 acc = make_float2(0.f, 0.f);
#pragma unroll 8
        for (int s2 = 0; s2 < S_; s2 += 2) {
            float2 tt = *reinterpret_cast<const float2*>(trow + s2);
            float2 ee = *reinterpret_cast<const float2*>(eh + s2);
            acc.x += tt.x * ee.x;
            acc.y += tt.y * ee.y;
        }
        float p = acc.x + acc.y;
        out_hid[t * (S_ * C_) + tid] = logf(p);
        expHt[(t * C_ + c) * S_ + sp] = (_Float16)p;
        __syncthreads();
        ehc[c * 260 + sp] = p;
        __syncthreads();
    }
}

// ---------------------------------------------------------------------------
// WMMA GEMM + fused chain combine:
//   P[n, t*4+c] = sum_s expE[s][n] * expH_t[s][c]   (f16 in, f32 acc)
//   log_obs[t][n] = log( sum_c wexp[c] * P[n, t*4+c] )
// One wave per 16x16 output tile; K=256 in 8 chunks of 32. Operands are
// stored K-contiguous per lane, so fragments load as 16B b128 vectors.
__global__ void ifhmm_obs_gemm(const _Float16* __restrict__ expEt,
                               const _Float16* __restrict__ expHt,
                               const float* __restrict__ wexp,
                               float* __restrict__ out_obs) {
    __shared__ float ldsP[16 * 16];
    int tn   = blockIdx.x;          // n tile (0..62)
    int ttc  = blockIdx.y;          // t*4+c tile (0..49)
    int lane = threadIdx.x;         // wave32
    int lo16 = lane & 15;
    bool hi  = lane >= 16;
    int m    = tn * 16 + lo16;      // n row of expEt (pad rows are 0)
    int ncol = ttc * 16 + lo16;     // tc row of expHt

    const h8* pa = reinterpret_cast<const h8*>(expEt + m * S_);
    const h8* pb = reinterpret_cast<const h8*>(expHt + ncol * S_);

    v8f acc = {};
    for (int k0 = 0; k0 < S_; k0 += 32) {
        // A fragment: lane holds K = k0+{0..7,16..23} (lo) / {8..15,24..31} (hi)
        h8 a0 = pa[(k0 >> 3) + (hi ? 1 : 0)];
        h8 a1 = pa[(k0 >> 3) + (hi ? 3 : 2)];
        v16h a = __builtin_shufflevector(a0, a1, 0, 1, 2, 3, 4, 5, 6, 7,
                                         8, 9, 10, 11, 12, 13, 14, 15);
        // B fragment: lane holds K = k0+{0..7,16..23} (lo) / {8..15,24..31} (hi)
        h8 b0 = pb[(k0 >> 3) + (hi ? 1 : 0)];
        h8 b1 = pb[(k0 >> 3) + (hi ? 3 : 2)];
        v16h b = __builtin_shufflevector(b0, b1, 0, 1, 2, 3, 4, 5, 6, 7,
                                         8, 9, 10, 11, 12, 13, 14, 15);
        acc = __builtin_amdgcn_wmma_f32_16x16x32_f16(
            /*neg_a=*/false, a, /*neg_b=*/false, b,
            /*c_mod=*/(short)0, acc, /*reuse_a=*/false, /*reuse_b=*/false);
    }
    // D layout: element r -> row M = r + (hi?8:0), col N = lane&15.
#pragma unroll
    for (int r = 0; r < 8; ++r)
        ldsP[(r + (hi ? 8 : 0)) * 16 + lo16] = acc[r];
    __syncthreads();

    float w0 = wexp[0], w1 = wexp[1], w2 = wexp[2], w3 = wexp[3];
    for (int idx = lane; idx < 64; idx += 32) {
        int row = idx & 15;         // n within tile
        int tq  = idx >> 4;         // which of 4 timesteps in this 16-col tile
        int n   = tn * 16 + row;
        int t   = ttc * 4 + tq;
        const float* pr = &ldsP[row * 16 + tq * 4];
        if (n < N_) {
            float ssum = pr[0] * w0 + pr[1] * w1 + pr[2] * w2 + pr[3] * w3;
            out_obs[(size_t)t * N_ + n] = logf(ssum);
        }
    }
}

// ---------------------------------------------------------------------------
// log_obs_[t,s,c,n] = log_emission[s,n] + log_hidden[t,s,c].
// One block per (t,s): the emission float4 is loaded once, reused for all 4
// chains; 819 MB written with non-temporal hints (output >> 192MB L2).
__global__ void ifhmm_obs_full(const float* __restrict__ out_emi,
                               const float* __restrict__ out_hid,
                               float* __restrict__ out_obsf) {
    int b = blockIdx.x;             // == t*256 + s
    int s = b & (S_ - 1);
    int tid = threadIdx.x;
    if (tid >= N_ / 4) return;
    const f4* er = reinterpret_cast<const f4*>(out_emi + (size_t)s * N_);
    f4 e = er[tid];
    const float* hp = out_hid + (size_t)b * C_;
    f4* orow = reinterpret_cast<f4*>(out_obsf + (size_t)b * C_ * N_) + tid;
#pragma unroll
    for (int c = 0; c < C_; ++c) {
        f4 o = e + hp[c];
        __builtin_nontemporal_store(o, orow);
        orow += N_ / 4;
    }
}

// ---------------------------------------------------------------------------
extern "C" void kernel_launch(void* const* d_in, const int* in_sizes, int n_in,
                              void* d_out, int out_size, void* d_ws, size_t ws_size,
                              hipStream_t stream) {
    const float* usi = (const float*)d_in[0];   // [S,C]
    const float* ucw = (const float*)d_in[1];   // [C]
    const float* ue  = (const float*)d_in[2];   // [S,N]
    const float* ut  = (const float*)d_in[3];   // [S,S]
    // d_in[4] = num_iters (fixed at 200 by the reference)

    float* out = (float*)d_out;
    float* out_obs  = out + OUT_OBS;
    float* out_obsf = out + OUT_OBSF;
    float* out_hid  = out + OUT_HID;
    float* out_emi  = out + OUT_EMI;
    float* out_w    = out + OUT_W;

    char* ws = (char*)d_ws;
    float*    expTtr = (float*)   (ws + WS_EXPTTR);
    _Float16* expEt  = (_Float16*)(ws + WS_EXPET);
    _Float16* expHt  = (_Float16*)(ws + WS_EXPHT);
    float*    wexp   = (float*)   (ws + WS_WEXP);

    // Normalizations (independent).
    ifhmm_norm_small<<<1, 256, 0, stream>>>(usi, ucw, out_hid, out_w, wexp,
                                            expHt, expEt);
    ifhmm_norm_transition<<<S_, 256, 0, stream>>>(ut, expTtr);
    ifhmm_norm_emission<<<S_, 256, 0, stream>>>(ue, out_emi, expEt);

    // Sequential forward recursion on one WGP.
    ifhmm_forward_recursion<<<1, 1024, 0, stream>>>(expTtr, out_hid, expHt);

    // WMMA GEMM: 63 n-tiles x 50 tc-tiles, one wave each.
    ifhmm_obs_gemm<<<dim3(NP_ / 16, TC_ / 16), 32, 0, stream>>>(expEt, expHt,
                                                                wexp, out_obs);

    // Bandwidth-dominated 819 MB broadcast-add, NT stores.
    ifhmm_obs_full<<<T_ * S_, 256, 0, stream>>>(out_emi, out_hid, out_obsf);
}
